// rNet_4690104287771
// MI455X (gfx1250) — compile-verified, matrix operand order checked by reading, not run on hardware
//
#include <hip/hip_runtime.h>
#include <hip/hip_bf16.h>

// ---------------------------------------------------------------------------
// R-Net forward for MI455X (gfx1250, wave32, WMMA 16x16x32 bf16).
// All matmuls use v_wmma_f32_16x16x32_bf16 with batch-8 padded to M=16.
// Recurrent state lives in LDS; weights stream from global (L2-resident).
// K-outer / accumulator-resident loops: one A fragment feeds 3-4 WMMAs.
// ---------------------------------------------------------------------------

typedef __attribute__((ext_vector_type(16))) __bf16 v16bf;
typedef __attribute__((ext_vector_type(8)))  __bf16 v8bf;
typedef __attribute__((ext_vector_type(8)))  float  v8f;
typedef __attribute__((ext_vector_type(4)))  float  v4f;

#define WMMA_BF16(a, b, c) \
  __builtin_amdgcn_wmma_f32_16x16x32_bf16(false, (a), false, (b), (short)0, (c), false, false)

__device__ __forceinline__ float sigf(float x) { return 1.f / (1.f + __expf(-x)); }

// --- K-major fragments (A-row / B-col index = lane&15). --------------------
// ISA 7.12.2: lanes 0-15 hold K [0..7],[16..23]; lanes 16-31 hold +8.
// Pointer-based loaders: p already includes row*ld + ((lane>>4)<<3).
__device__ __forceinline__ v16bf frag_bf16_p(const __bf16* p) {
  v8bf lo = *(const v8bf*)p;
  v8bf hi = *(const v8bf*)(p + 16);
  v16bf r;
#pragma unroll
  for (int i = 0; i < 8; ++i) { r[i] = lo[i]; r[8 + i] = hi[i]; }
  return r;
}

__device__ __forceinline__ v16bf frag_f32_p(const float* p) {
  v4f a0 = *(const v4f*)p;
  v4f a1 = *(const v4f*)(p + 4);
  v4f b0 = *(const v4f*)(p + 16);
  v4f b1 = *(const v4f*)(p + 20);
  v16bf out;
#pragma unroll
  for (int i = 0; i < 4; ++i) {
    out[i]      = (__bf16)a0[i];
    out[4 + i]  = (__bf16)a1[i];
    out[8 + i]  = (__bf16)b0[i];
    out[12 + i] = (__bf16)b1[i];
  }
  return out;
}

__device__ __forceinline__ v16bf frag_bf16(const __bf16* base, int ld, int r0, int kbase) {
  int lane = threadIdx.x & 31;
  return frag_bf16_p(base + (size_t)(r0 + (lane & 15)) * ld + kbase + ((lane >> 4) << 3));
}

// Unguarded: caller guarantees 16 valid (possibly zero-padded) rows.
__device__ __forceinline__ v16bf frag_f32(const float* base, int ld, int r0, int kbase) {
  int lane = threadIdx.x & 31;
  return frag_f32_p(base + (size_t)(r0 + (lane & 15)) * ld + kbase + ((lane >> 4) << 3));
}

// C tile rows 0..7 (the real batch) -> LDS [8][ld]. Lanes 0-15 hold M=0..7.
__device__ __forceinline__ void tile_rows8(float* dst, int ld, int n0, v8f acc,
                                           const float* bias0, const float* bias1) {
  int lane = threadIdx.x & 31;
  if (lane < 16) {
    int n = n0 + lane;
    float bb = (bias0 ? bias0[n] : 0.f) + (bias1 ? bias1[n] : 0.f);
#pragma unroll
    for (int i = 0; i < 8; ++i) dst[(size_t)i * ld + n] = acc[i] + bb;
  }
}

// Full 16-row C tile to global with row guard.
__device__ __forceinline__ void tile_store_g(float* C, int ldc, int m0, int M, int n0, v8f acc,
                                             const float* bias) {
  int lane = threadIdx.x & 31;
  int n = n0 + (lane & 15);
  int mb = m0 + ((lane >> 4) << 3);
  float bb = bias ? bias[n] : 0.f;
#pragma unroll
  for (int i = 0; i < 8; ++i) {
    int m = mb + i;
    if (m < M) C[(size_t)m * ldc + n] = acc[i] + bb;
  }
}

// ---------------------------------------------------------------------------
// Weight convert f32 -> bf16 with K padding to a multiple of 32.
__global__ void k_cvt_pad(const float* __restrict__ src, __bf16* __restrict__ dst, int N, int K,
                          int ldk) {
  size_t total = (size_t)N * ldk;
  for (size_t i = (size_t)blockIdx.x * blockDim.x + threadIdx.x; i < total;
       i += (size_t)gridDim.x * blockDim.x) {
    int k = (int)(i % ldk);
    int n = (int)(i / ldk);
    dst[i] = (k < K) ? (__bf16)src[(size_t)n * K + k] : (__bf16)0.f;
  }
}

// Word embedding gather into X[s][0..299] (f32, ld=832), zero pad cols 812..831.
__global__ void k_words(const int* __restrict__ ids, const float* __restrict__ Wemb,
                        float* __restrict__ X, int S) {
  int total = S * 320;
  for (int i = blockIdx.x * blockDim.x + threadIdx.x; i < total; i += gridDim.x * blockDim.x) {
    int s = i / 320, c = i % 320;
    if (c < 300)
      X[(size_t)s * 832 + c] = Wemb[(size_t)ids[s] * 300 + c];
    else
      X[(size_t)s * 832 + 812 + (c - 300)] = 0.f;
  }
}

// ---------------------------------------------------------------------------
// Generic GEMM: C[M,N] = A[M,K]f32 @ B[N,K]bf16^T (+bias).
// 8 waves/block, exactly 4 N-tiles per active wave (N % 64 == 0 for every
// launch: 768/256), K-outer so each A fragment feeds 4 WMMAs. Named
// accumulators (no arrays) -> no v_movrel VGPR indexing. M % 16 == 0.
__global__ __launch_bounds__(256) void k_gemm(const float* __restrict__ A, int lda,
                                              const __bf16* __restrict__ B, int ldb,
                                              const float* __restrict__ bias,
                                              float* __restrict__ C, int ldc, int M, int N,
                                              int Kt) {
  int wave = threadIdx.x >> 5;
  int lane = threadIdx.x & 31;
  int m0 = blockIdx.x * 16;
  int n0 = (blockIdx.y * 8 + wave) * 64;
  if (n0 + 64 > N) return;
  // Per-lane base pointers; inner loop address math is just +kt*32.
  const float* Ap = A + (size_t)(m0 + (lane & 15)) * lda + ((lane >> 4) << 3);
  const __bf16* B0 = B + (size_t)(n0 + (lane & 15)) * ldb + ((lane >> 4) << 3);
  const __bf16* B1 = B0 + (size_t)16 * ldb;
  const __bf16* B2 = B0 + (size_t)32 * ldb;
  const __bf16* B3 = B0 + (size_t)48 * ldb;
  v8f acc0 = {}, acc1 = {}, acc2 = {}, acc3 = {};
  for (int kt = 0; kt < Kt; ++kt) {
    int ko = kt * 32;
    v16bf a = frag_f32_p(Ap + ko);
    if (kt + 1 < Kt)  // L2-hit latency hiding: global_prefetch of next K-tile
      __builtin_prefetch(B0 + ko + 32, 0, 1);
    acc0 = WMMA_BF16(a, frag_bf16_p(B0 + ko), acc0);
    acc1 = WMMA_BF16(a, frag_bf16_p(B1 + ko), acc1);
    acc2 = WMMA_BF16(a, frag_bf16_p(B2 + ko), acc2);
    acc3 = WMMA_BF16(a, frag_bf16_p(B3 + ko), acc3);
  }
  tile_store_g(C, ldc, m0, M, n0, acc0, bias);
  tile_store_g(C, ldc, m0, M, n0 + 16, acc1, bias);
  tile_store_g(C, ldc, m0, M, n0 + 32, acc2, bias);
  tile_store_g(C, ldc, m0, M, n0 + 48, acc3, bias);
}

// ---------------------------------------------------------------------------
// Char GRU: each WG owns 8 independent sequences (12 chars, fwd+bwd),
// writes [hf|hb] (512 floats) into X cols 300..811.
__global__ __launch_bounds__(512) void k_char_gru(const int* __restrict__ chars, int S,
                                                  const float* __restrict__ Cemb,
                                                  const __bf16* __restrict__ Wih,   // [768,160]
                                                  const __bf16* __restrict__ Whh,   // [768,256]
                                                  const float* __restrict__ bih,
                                                  const float* __restrict__ bhh,
                                                  float* __restrict__ X) {          // [S,832]
  __shared__ __bf16 xb[16][12 * 160];  // rows 8..15 zero (WMMA M padding)
  __shared__ float h[16][256];
  __shared__ float gi[8 * 768];
  __shared__ float gh[8 * 768];
  int s0 = blockIdx.x * 8;
  int tid = threadIdx.x, wave = tid >> 5;

  for (int i = tid; i < 16 * 12 * 160; i += 512) {
    int r = i / 1920, rem = i % 1920, w = rem / 160, k = rem % 160;
    __bf16 v = (__bf16)0.f;
    int s = s0 + r;
    if (r < 8 && s < S && k < 150) v = (__bf16)Cemb[(size_t)chars[s * 12 + w] * 150 + k];
    xb[r][rem] = v;
  }
  for (int i = tid; i < 16 * 256; i += 512) (&h[0][0])[i] = 0.f;
  __syncthreads();

  for (int dir = 0; dir < 2; ++dir) {
    if (dir == 1) {
      for (int i = tid; i < 8 * 256; i += 512) h[i >> 8][i & 255] = 0.f;
      __syncthreads();
    }
    for (int st = 0; st < 12; ++st) {
      int w = dir ? (11 - st) : st;
      // 48 N-tiles = 16 waves x 3; K-outer, accumulators resident.
      v8f ai0 = {}, ai1 = {}, ai2 = {}, ah0 = {}, ah1 = {}, ah2 = {};
#pragma unroll
      for (int kt = 0; kt < 5; ++kt) {
        v16bf a = frag_bf16(&xb[0][w * 160], 1920, 0, kt * 32);
        ai0 = WMMA_BF16(a, frag_bf16(Wih, 160, (wave * 3 + 0) * 16, kt * 32), ai0);
        ai1 = WMMA_BF16(a, frag_bf16(Wih, 160, (wave * 3 + 1) * 16, kt * 32), ai1);
        ai2 = WMMA_BF16(a, frag_bf16(Wih, 160, (wave * 3 + 2) * 16, kt * 32), ai2);
      }
#pragma unroll
      for (int kt = 0; kt < 8; ++kt) {
        v16bf a = frag_f32(&h[0][0], 256, 0, kt * 32);
        ah0 = WMMA_BF16(a, frag_bf16(Whh, 256, (wave * 3 + 0) * 16, kt * 32), ah0);
        ah1 = WMMA_BF16(a, frag_bf16(Whh, 256, (wave * 3 + 1) * 16, kt * 32), ah1);
        ah2 = WMMA_BF16(a, frag_bf16(Whh, 256, (wave * 3 + 2) * 16, kt * 32), ah2);
      }
      tile_rows8(gi, 768, (wave * 3 + 0) * 16, ai0, bih, nullptr);
      tile_rows8(gi, 768, (wave * 3 + 1) * 16, ai1, bih, nullptr);
      tile_rows8(gi, 768, (wave * 3 + 2) * 16, ai2, bih, nullptr);
      tile_rows8(gh, 768, (wave * 3 + 0) * 16, ah0, bhh, nullptr);
      tile_rows8(gh, 768, (wave * 3 + 1) * 16, ah1, bhh, nullptr);
      tile_rows8(gh, 768, (wave * 3 + 2) * 16, ah2, bhh, nullptr);
      __syncthreads();
      for (int i = tid; i < 8 * 256; i += 512) {
        int b = i >> 8, j = i & 255;
        float r = sigf(gi[b * 768 + j] + gh[b * 768 + j]);
        float z = sigf(gi[b * 768 + 256 + j] + gh[b * 768 + 256 + j]);
        float n = tanhf(gi[b * 768 + 512 + j] + r * gh[b * 768 + 512 + j]);
        h[b][j] = (1.f - z) * n + z * h[b][j];
      }
      __syncthreads();
    }
    for (int i = tid; i < 8 * 256; i += 512) {
      int r = i >> 8, j = i & 255;
      int s = s0 + r;
      if (s < S) X[(size_t)s * 832 + 300 + dir * 256 + j] = h[r][j];
    }
    __syncthreads();
  }
}

// ---------------------------------------------------------------------------
// Sequence GRU (batch 8). blockIdx.x = direction. Gi (incl. bih) precomputed.
__global__ __launch_bounds__(512) void k_gru_seq(const float* __restrict__ Gi, size_t gi_dstride,
                                                 const __bf16* __restrict__ Whh,  // [768,256]
                                                 const float* __restrict__ bhh,
                                                 float* __restrict__ Hout, size_t out_dstride,
                                                 int ldo, int dircol, int T) {
  int dir = blockIdx.x;
  Gi += (size_t)dir * gi_dstride;
  Hout += (size_t)dir * out_dstride;
  int col0 = dir * dircol;
  __shared__ float h[16][256];
  __shared__ float gh[8 * 768];
  int tid = threadIdx.x, wave = tid >> 5;
  for (int i = tid; i < 16 * 256; i += 512) (&h[0][0])[i] = 0.f;
  __syncthreads();
  for (int s = 0; s < T; ++s) {
    int t = dir ? (T - 1 - s) : s;
    v8f a0 = {}, a1 = {}, a2 = {};
#pragma unroll
    for (int kt = 0; kt < 8; ++kt) {
      v16bf a = frag_f32(&h[0][0], 256, 0, kt * 32);
      a0 = WMMA_BF16(a, frag_bf16(Whh, 256, (wave * 3 + 0) * 16, kt * 32), a0);
      a1 = WMMA_BF16(a, frag_bf16(Whh, 256, (wave * 3 + 1) * 16, kt * 32), a1);
      a2 = WMMA_BF16(a, frag_bf16(Whh, 256, (wave * 3 + 2) * 16, kt * 32), a2);
    }
    tile_rows8(gh, 768, (wave * 3 + 0) * 16, a0, bhh, nullptr);
    tile_rows8(gh, 768, (wave * 3 + 1) * 16, a1, bhh, nullptr);
    tile_rows8(gh, 768, (wave * 3 + 2) * 16, a2, bhh, nullptr);
    __syncthreads();
    const float* gi = Gi + (size_t)t * 8 * 768;
    for (int i = tid; i < 8 * 256; i += 512) {
      int b = i >> 8, j = i & 255;
      float r = sigf(gi[b * 768 + j] + gh[b * 768 + j]);
      float z = sigf(gi[b * 768 + 256 + j] + gh[b * 768 + 256 + j]);
      float n = tanhf(gi[b * 768 + 512 + j] + r * gh[b * 768 + 512 + j]);
      float hn = (1.f - z) * n + z * h[b][j];
      h[b][j] = hn;
      Hout[(size_t)(t * 8 + b) * ldo + col0 + j] = hn;
    }
    __syncthreads();
  }
}

// ---------------------------------------------------------------------------
// Attention-gated LSTM scan (match-LSTM / self-matching). blockIdx.x = dir.
__global__ __launch_bounds__(512) void k_attn_lstm(
    const float* __restrict__ pre,   // [M*8,256]
    const float* __restrict__ hinW,  // [T*8,256] or nullptr
    const float* __restrict__ Hmem,  // [M*8,512]
    const float* __restrict__ Hin,   // [T*8,512]
    const __bf16* __restrict__ Wh,   // [256,256]
    const float* __restrict__ wsc,   // [256]
    const __bf16* __restrict__ Wg,   // [1024,1024]
    const __bf16* __restrict__ Wih,  // [1024,1024]
    const __bf16* __restrict__ Whh,  // [1024,256]
    const float* __restrict__ bih, const float* __restrict__ bhh,
    float* __restrict__ Hout, int ldo, int dircol, int T, int M) {
  int dir = blockIdx.x;
  int col0 = dir * dircol;
  __shared__ float h[16][256];
  __shared__ float cc[8][256];
  __shared__ float hW[8 * 256];
  __shared__ float sc[2048];
  __shared__ float red[16];
  __shared__ __bf16 z[16][1024];
  __shared__ float g[8 * 1024];
  int tid = threadIdx.x, lane = tid & 31, wave = tid >> 5;
  for (int i = tid; i < 16 * 256; i += 512) (&h[0][0])[i] = 0.f;
  for (int i = tid; i < 8 * 256; i += 512) (&cc[0][0])[i] = 0.f;
  for (int i = tid; i < 16 * 1024; i += 512) (&z[0][0])[i] = (__bf16)0.f;
  __syncthreads();

  for (int s = 0; s < T; ++s) {
    int t = dir ? (T - 1 - s) : s;
    {  // hW = h @ Wh^T  (16 tiles, one per wave)
      v8f acc = {};
#pragma unroll
      for (int kt = 0; kt < 8; ++kt) {
        v16bf a = frag_f32(&h[0][0], 256, 0, kt * 32);
        acc = WMMA_BF16(a, frag_bf16(Wh, 256, wave * 16, kt * 32), acc);
      }
      tile_rows8(hW, 256, wave * 16, acc, nullptr, nullptr);
    }
    __syncthreads();
    // scores[m,b] = w . tanh(pre + hW (+ hinW))
    for (int i = tid; i < M * 8; i += 512) {
      int b = i & 7;
      const float* pp = pre + (size_t)i * 256;
      const float* hv = hinW ? hinW + (size_t)(t * 8 + b) * 256 : nullptr;
      const float* hw = hW + b * 256;
      float a = 0.f;
      for (int j = 0; j < 256; j += 4) {
        v4f p4 = *(const v4f*)(pp + j);
        v4f w4 = *(const v4f*)(wsc + j);
#pragma unroll
        for (int u = 0; u < 4; ++u) {
          float v = p4[u] + hw[j + u] + (hv ? hv[j + u] : 0.f);
          a += w4[u] * tanhf(v);
        }
      }
      sc[i] = a;
    }
    __syncthreads();
    // softmax over m per batch column (wave b reduces with shuffles)
    if (wave < 8) {
      int b = wave;
      float mx = -3.4e38f;
      for (int m = lane; m < M; m += 32) mx = fmaxf(mx, sc[m * 8 + b]);
#pragma unroll
      for (int o = 16; o > 0; o >>= 1) mx = fmaxf(mx, __shfl_xor(mx, o, 32));
      float sm = 0.f;
      for (int m = lane; m < M; m += 32) sm += __expf(sc[m * 8 + b] - mx);
#pragma unroll
      for (int o = 16; o > 0; o >>= 1) sm += __shfl_xor(sm, o, 32);
      if (lane == 0) { red[b] = mx; red[8 + b] = 1.f / sm; }
    }
    __syncthreads();
    for (int i = tid; i < M * 8; i += 512) {
      int b = i & 7;
      sc[i] = __expf(sc[i] - red[b]) * red[8 + b];
    }
    __syncthreads();
    // ctx + build z = [hin_t | ctx]
    for (int i = tid; i < 8 * 512; i += 512) {
      int b = i >> 9, d = i & 511;
      float a = 0.f;
      for (int m = 0; m < M; ++m) a += sc[m * 8 + b] * Hmem[((size_t)m * 8 + b) * 512 + d];
      z[b][512 + d] = (__bf16)a;
      z[b][d] = (__bf16)Hin[(size_t)(t * 8 + b) * 512 + d];
    }
    __syncthreads();
    // gate = z @ Wg^T  (64 tiles = 16 waves x 4, K-outer)
    {
      v8f c0 = {}, c1 = {}, c2 = {}, c3 = {};
      for (int kt = 0; kt < 32; ++kt) {
        v16bf a = frag_bf16(&z[0][0], 1024, 0, kt * 32);
        c0 = WMMA_BF16(a, frag_bf16(Wg, 1024, (wave * 4 + 0) * 16, kt * 32), c0);
        c1 = WMMA_BF16(a, frag_bf16(Wg, 1024, (wave * 4 + 1) * 16, kt * 32), c1);
        c2 = WMMA_BF16(a, frag_bf16(Wg, 1024, (wave * 4 + 2) * 16, kt * 32), c2);
        c3 = WMMA_BF16(a, frag_bf16(Wg, 1024, (wave * 4 + 3) * 16, kt * 32), c3);
      }
      tile_rows8(g, 1024, (wave * 4 + 0) * 16, c0, nullptr, nullptr);
      tile_rows8(g, 1024, (wave * 4 + 1) * 16, c1, nullptr, nullptr);
      tile_rows8(g, 1024, (wave * 4 + 2) * 16, c2, nullptr, nullptr);
      tile_rows8(g, 1024, (wave * 4 + 3) * 16, c3, nullptr, nullptr);
    }
    __syncthreads();
    for (int i = tid; i < 8 * 1024; i += 512) {
      int b = i >> 10, j = i & 1023;
      float zz = (float)z[b][j];
      z[b][j] = (__bf16)(zz * sigf(g[i]));
    }
    __syncthreads();
    // LSTM gates = zg @ Wih^T + h @ Whh^T + bih + bhh
    {
      v8f c0 = {}, c1 = {}, c2 = {}, c3 = {};
      for (int kt = 0; kt < 32; ++kt) {
        v16bf a = frag_bf16(&z[0][0], 1024, 0, kt * 32);
        c0 = WMMA_BF16(a, frag_bf16(Wih, 1024, (wave * 4 + 0) * 16, kt * 32), c0);
        c1 = WMMA_BF16(a, frag_bf16(Wih, 1024, (wave * 4 + 1) * 16, kt * 32), c1);
        c2 = WMMA_BF16(a, frag_bf16(Wih, 1024, (wave * 4 + 2) * 16, kt * 32), c2);
        c3 = WMMA_BF16(a, frag_bf16(Wih, 1024, (wave * 4 + 3) * 16, kt * 32), c3);
      }
#pragma unroll
      for (int kt = 0; kt < 8; ++kt) {
        v16bf a = frag_f32(&h[0][0], 256, 0, kt * 32);
        c0 = WMMA_BF16(a, frag_bf16(Whh, 256, (wave * 4 + 0) * 16, kt * 32), c0);
        c1 = WMMA_BF16(a, frag_bf16(Whh, 256, (wave * 4 + 1) * 16, kt * 32), c1);
        c2 = WMMA_BF16(a, frag_bf16(Whh, 256, (wave * 4 + 2) * 16, kt * 32), c2);
        c3 = WMMA_BF16(a, frag_bf16(Whh, 256, (wave * 4 + 3) * 16, kt * 32), c3);
      }
      tile_rows8(g, 1024, (wave * 4 + 0) * 16, c0, bih, bhh);
      tile_rows8(g, 1024, (wave * 4 + 1) * 16, c1, bih, bhh);
      tile_rows8(g, 1024, (wave * 4 + 2) * 16, c2, bih, bhh);
      tile_rows8(g, 1024, (wave * 4 + 3) * 16, c3, bih, bhh);
    }
    __syncthreads();
    for (int i = tid; i < 8 * 256; i += 512) {
      int b = i >> 8, j = i & 255;
      float ig = sigf(g[b * 1024 + j]);
      float fg = sigf(g[b * 1024 + 256 + j]);
      float gg = tanhf(g[b * 1024 + 512 + j]);
      float og = sigf(g[b * 1024 + 768 + j]);
      float cn = fg * cc[b][j] + ig * gg;
      cc[b][j] = cn;
      float hn = og * tanhf(cn);
      h[b][j] = hn;
      Hout[(size_t)(t * 8 + b) * ldo + col0 + j] = hn;
    }
    __syncthreads();
  }
}

// ---------------------------------------------------------------------------
// Answer pointer: 2 softmax steps over P=256, batch 8. Writes d_out [2,256,8].
__global__ __launch_bounds__(512) void k_ans(const float* __restrict__ pre,   // [2048,256]
                                             const float* __restrict__ Hs,    // [2048,512]
                                             const __bf16* __restrict__ Wans, // [256,256]
                                             const float* __restrict__ bw,    // [256]
                                             const __bf16* __restrict__ Wih,  // [1024,512]
                                             const __bf16* __restrict__ Whh,  // [1024,256]
                                             const float* __restrict__ bih,
                                             const float* __restrict__ bhh,
                                             float* __restrict__ out) {
  __shared__ float h[16][256];
  __shared__ float cc[8][256];
  __shared__ float hW[8 * 256];
  __shared__ float sc[2048];
  __shared__ float red[16];
  __shared__ __bf16 z[16][512];
  __shared__ float g[8 * 1024];
  int tid = threadIdx.x, lane = tid & 31, wave = tid >> 5;
  const int M = 256;
  for (int i = tid; i < 16 * 256; i += 512) (&h[0][0])[i] = 0.f;
  for (int i = tid; i < 8 * 256; i += 512) (&cc[0][0])[i] = 0.f;
  for (int i = tid; i < 16 * 512; i += 512) (&z[0][0])[i] = (__bf16)0.f;
  __syncthreads();

  for (int step = 0; step < 2; ++step) {
    {  // hW = h @ Wans^T
      v8f acc = {};
#pragma unroll
      for (int kt = 0; kt < 8; ++kt) {
        v16bf a = frag_f32(&h[0][0], 256, 0, kt * 32);
        acc = WMMA_BF16(a, frag_bf16(Wans, 256, wave * 16, kt * 32), acc);
      }
      tile_rows8(hW, 256, wave * 16, acc, nullptr, nullptr);
    }
    __syncthreads();
    for (int i = tid; i < M * 8; i += 512) {
      int b = i & 7;
      const float* pp = pre + (size_t)i * 256;
      const float* hw = hW + b * 256;
      float a = 0.f;
      for (int j = 0; j < 256; j += 4) {
        v4f p4 = *(const v4f*)(pp + j);
        v4f w4 = *(const v4f*)(bw + j);
#pragma unroll
        for (int u = 0; u < 4; ++u) a += w4[u] * tanhf(p4[u] + hw[j + u]);
      }
      sc[i] = a;
    }
    __syncthreads();
    if (wave < 8) {
      int b = wave;
      float mx = -3.4e38f;
      for (int m = lane; m < M; m += 32) mx = fmaxf(mx, sc[m * 8 + b]);
#pragma unroll
      for (int o = 16; o > 0; o >>= 1) mx = fmaxf(mx, __shfl_xor(mx, o, 32));
      float sm = 0.f;
      for (int m = lane; m < M; m += 32) sm += __expf(sc[m * 8 + b] - mx);
#pragma unroll
      for (int o = 16; o > 0; o >>= 1) sm += __shfl_xor(sm, o, 32);
      if (lane == 0) { red[b] = mx; red[8 + b] = 1.f / sm; }
    }
    __syncthreads();
    for (int i = tid; i < M * 8; i += 512) {
      int b = i & 7;
      float a = __expf(sc[i] - red[b]) * red[8 + b];
      sc[i] = a;
      out[step * 2048 + i] = a;
    }
    __syncthreads();
    // ctx -> z rows 0..7
    for (int i = tid; i < 8 * 512; i += 512) {
      int b = i >> 9, d = i & 511;
      float a = 0.f;
      for (int m = 0; m < M; ++m) a += sc[m * 8 + b] * Hs[((size_t)m * 8 + b) * 512 + d];
      z[b][d] = (__bf16)a;
    }
    __syncthreads();
    {
      v8f c0 = {}, c1 = {}, c2 = {}, c3 = {};
#pragma unroll
      for (int kt = 0; kt < 16; ++kt) {
        v16bf a = frag_bf16(&z[0][0], 512, 0, kt * 32);
        c0 = WMMA_BF16(a, frag_bf16(Wih, 512, (wave * 4 + 0) * 16, kt * 32), c0);
        c1 = WMMA_BF16(a, frag_bf16(Wih, 512, (wave * 4 + 1) * 16, kt * 32), c1);
        c2 = WMMA_BF16(a, frag_bf16(Wih, 512, (wave * 4 + 2) * 16, kt * 32), c2);
        c3 = WMMA_BF16(a, frag_bf16(Wih, 512, (wave * 4 + 3) * 16, kt * 32), c3);
      }
#pragma unroll
      for (int kt = 0; kt < 8; ++kt) {
        v16bf a = frag_f32(&h[0][0], 256, 0, kt * 32);
        c0 = WMMA_BF16(a, frag_bf16(Whh, 256, (wave * 4 + 0) * 16, kt * 32), c0);
        c1 = WMMA_BF16(a, frag_bf16(Whh, 256, (wave * 4 + 1) * 16, kt * 32), c1);
        c2 = WMMA_BF16(a, frag_bf16(Whh, 256, (wave * 4 + 2) * 16, kt * 32), c2);
        c3 = WMMA_BF16(a, frag_bf16(Whh, 256, (wave * 4 + 3) * 16, kt * 32), c3);
      }
      tile_rows8(g, 1024, (wave * 4 + 0) * 16, c0, bih, bhh);
      tile_rows8(g, 1024, (wave * 4 + 1) * 16, c1, bih, bhh);
      tile_rows8(g, 1024, (wave * 4 + 2) * 16, c2, bih, bhh);
      tile_rows8(g, 1024, (wave * 4 + 3) * 16, c3, bih, bhh);
    }
    __syncthreads();
    for (int i = tid; i < 8 * 256; i += 512) {
      int b = i >> 8, j = i & 255;
      float ig = sigf(g[b * 1024 + j]);
      float fg = sigf(g[b * 1024 + 256 + j]);
      float gg = tanhf(g[b * 1024 + 512 + j]);
      float og = sigf(g[b * 1024 + 768 + j]);
      float cn = fg * cc[b][j] + ig * gg;
      cc[b][j] = cn;
      h[b][j] = og * tanhf(cn);
    }
    __syncthreads();
  }
}

// ---------------------------------------------------------------------------
// Host-side orchestration.
// Assumed input order (setup_inputs dict order, params in insertion order):
enum {
  IN_PID = 0, IN_QID, IN_CHP, IN_CHQ,
  IN_WEMB, IN_CEMB,
  IN_CG_WIH, IN_CG_WHH, IN_CG_BIH, IN_CG_BHH,
  IN_P0_WIH, IN_P0_WHH, IN_P0_BIH, IN_P0_BHH,
  IN_P1_WIH, IN_P1_WHH, IN_P1_BIH, IN_P1_BHH,
  IN_P2_WIH, IN_P2_WHH, IN_P2_BIH, IN_P2_BHH,
  IN_ATTN_Q, IN_ATTN_P, IN_ATTN_H, IN_ALPHA,
  IN_GATE_M,
  IN_ML_WIH, IN_ML_WHH, IN_ML_BIH, IN_ML_BHH,
  IN_ATTN_SP, IN_ATTN_SH, IN_GAMMA, IN_GATE_S,
  IN_SL_WIH, IN_SL_WHH, IN_SL_BIH, IN_SL_BHH,
  IN_ATTN_SL, IN_ATTN_ANS, IN_BETA,
  IN_ANS_WIH, IN_ANS_WHH, IN_ANS_BIH, IN_ANS_BHH
};

static inline char* bump(char*& p, size_t bytes) {
  char* r = p;
  p += (bytes + 255) & ~(size_t)255;
  return r;
}

extern "C" void kernel_launch(void* const* d_in, const int* in_sizes, int n_in,
                              void* d_out, int out_size, void* d_ws, size_t ws_size,
                              hipStream_t stream) {
  (void)in_sizes; (void)n_in; (void)out_size; (void)ws_size;
  const int PT = 256 * 8;  // passage rows
  const int QT = 30 * 8;   // question rows
  char* wp = (char*)d_ws;

  // bf16 weights
  __bf16* cgWih = (__bf16*)bump(wp, (size_t)768 * 160 * 2);
  __bf16* cgWhh = (__bf16*)bump(wp, (size_t)768 * 256 * 2);
  __bf16* p0Wih = (__bf16*)bump(wp, (size_t)768 * 832 * 2);
  __bf16* p0Whh = (__bf16*)bump(wp, (size_t)768 * 256 * 2);
  __bf16* p1Wih = (__bf16*)bump(wp, (size_t)768 * 256 * 2);
  __bf16* p1Whh = (__bf16*)bump(wp, (size_t)768 * 256 * 2);
  __bf16* p2Wih = (__bf16*)bump(wp, (size_t)768 * 256 * 2);
  __bf16* p2Whh = (__bf16*)bump(wp, (size_t)768 * 256 * 2);
  __bf16* aQ    = (__bf16*)bump(wp, (size_t)256 * 512 * 2);
  __bf16* aP    = (__bf16*)bump(wp, (size_t)256 * 512 * 2);
  __bf16* aH    = (__bf16*)bump(wp, (size_t)256 * 256 * 2);
  __bf16* gM    = (__bf16*)bump(wp, (size_t)1024 * 1024 * 2);
  __bf16* mlWih = (__bf16*)bump(wp, (size_t)1024 * 1024 * 2);
  __bf16* mlWhh = (__bf16*)bump(wp, (size_t)1024 * 256 * 2);
  __bf16* aSP   = (__bf16*)bump(wp, (size_t)256 * 512 * 2);
  __bf16* aSH   = (__bf16*)bump(wp, (size_t)256 * 256 * 2);
  __bf16* gS    = (__bf16*)bump(wp, (size_t)1024 * 1024 * 2);
  __bf16* slWih = (__bf16*)bump(wp, (size_t)1024 * 1024 * 2);
  __bf16* slWhh = (__bf16*)bump(wp, (size_t)1024 * 256 * 2);
  __bf16* aSL   = (__bf16*)bump(wp, (size_t)256 * 512 * 2);
  __bf16* aANS  = (__bf16*)bump(wp, (size_t)256 * 256 * 2);
  __bf16* anWih = (__bf16*)bump(wp, (size_t)1024 * 512 * 2);
  __bf16* anWhh = (__bf16*)bump(wp, (size_t)1024 * 256 * 2);

  // f32 activations
  float* px    = (float*)bump(wp, (size_t)PT * 832 * 4);
  float* qx    = (float*)bump(wp, (size_t)QT * 832 * 4);
  float* GiP0  = (float*)bump(wp, (size_t)PT * 768 * 4);
  float* GiPd  = (float*)bump(wp, (size_t)2 * PT * 768 * 4);
  float* GiQ0  = (float*)bump(wp, (size_t)QT * 768 * 4);
  float* GiQd  = (float*)bump(wp, (size_t)2 * QT * 768 * 4);
  float* hsPa  = (float*)bump(wp, (size_t)2 * PT * 256 * 4);
  float* hsPb  = (float*)bump(wp, (size_t)2 * PT * 256 * 4);
  float* hsQa  = (float*)bump(wp, (size_t)2 * QT * 256 * 4);
  float* hsQb  = (float*)bump(wp, (size_t)2 * QT * 256 * 4);
  float* HpF   = (float*)bump(wp, (size_t)PT * 512 * 4);
  float* HqF   = (float*)bump(wp, (size_t)QT * 512 * 4);
  float* preQ  = (float*)bump(wp, (size_t)QT * 256 * 4);
  float* hinWP = (float*)bump(wp, (size_t)PT * 256 * 4);
  float* Hm    = (float*)bump(wp, (size_t)PT * 512 * 4);
  float* preS  = (float*)bump(wp, (size_t)PT * 256 * 4);
  float* Hs    = (float*)bump(wp, (size_t)PT * 512 * 4);
  float* preA  = (float*)bump(wp, (size_t)PT * 256 * 4);

  auto F  = [&](int i) { return (const float*)d_in[i]; };
  auto cvt = [&](int idx, __bf16* dst, int N, int K, int ldk) {
    size_t total = (size_t)N * ldk;
    int blocks = (int)((total + 255) / 256);
    if (blocks > 4096) blocks = 4096;
    k_cvt_pad<<<blocks, 256, 0, stream>>>(F(idx), dst, N, K, ldk);
  };

  cvt(IN_CG_WIH, cgWih, 768, 150, 160);
  cvt(IN_CG_WHH, cgWhh, 768, 256, 256);
  cvt(IN_P0_WIH, p0Wih, 768, 812, 832);
  cvt(IN_P0_WHH, p0Whh, 768, 256, 256);
  cvt(IN_P1_WIH, p1Wih, 768, 256, 256);
  cvt(IN_P1_WHH, p1Whh, 768, 256, 256);
  cvt(IN_P2_WIH, p2Wih, 768, 256, 256);
  cvt(IN_P2_WHH, p2Whh, 768, 256, 256);
  cvt(IN_ATTN_Q, aQ, 256, 512, 512);
  cvt(IN_ATTN_P, aP, 256, 512, 512);
  cvt(IN_ATTN_H, aH, 256, 256, 256);
  cvt(IN_GATE_M, gM, 1024, 1024, 1024);
  cvt(IN_ML_WIH, mlWih, 1024, 1024, 1024);
  cvt(IN_ML_WHH, mlWhh, 1024, 256, 256);
  cvt(IN_ATTN_SP, aSP, 256, 512, 512);
  cvt(IN_ATTN_SH, aSH, 256, 256, 256);
  cvt(IN_GATE_S, gS, 1024, 1024, 1024);
  cvt(IN_SL_WIH, slWih, 1024, 1024, 1024);
  cvt(IN_SL_WHH, slWhh, 1024, 256, 256);
  cvt(IN_ATTN_SL, aSL, 256, 512, 512);
  cvt(IN_ATTN_ANS, aANS, 256, 256, 256);
  cvt(IN_ANS_WIH, anWih, 1024, 512, 512);
  cvt(IN_ANS_WHH, anWhh, 1024, 256, 256);

  // embeddings + char GRU -> px/qx (f32, [rows,832])
  k_words<<<(PT * 320 + 255) / 256, 256, 0, stream>>>((const int*)d_in[IN_PID], F(IN_WEMB), px, PT);
  k_words<<<(QT * 320 + 255) / 256, 256, 0, stream>>>((const int*)d_in[IN_QID], F(IN_WEMB), qx, QT);
  k_char_gru<<<(PT + 7) / 8, 512, 0, stream>>>((const int*)d_in[IN_CHP], PT, F(IN_CEMB), cgWih,
                                               cgWhh, F(IN_CG_BIH), F(IN_CG_BHH), px);
  k_char_gru<<<(QT + 7) / 8, 512, 0, stream>>>((const int*)d_in[IN_CHQ], QT, F(IN_CEMB), cgWih,
                                               cgWhh, F(IN_CG_BIH), F(IN_CG_BHH), qx);

  auto gemm = [&](const float* A, int lda, const __bf16* B, int ldb, const float* bias, float* C,
                  int ldc, int Mm, int Nn, int Kpad) {
    dim3 gr((Mm + 15) / 16, (Nn + 511) / 512);
    k_gemm<<<gr, 256, 0, stream>>>(A, lda, B, ldb, bias, C, ldc, Mm, Nn, Kpad / 32);
  };

  // ---- passage 3-layer bi-GRU ----
  gemm(px, 832, p0Wih, 832, F(IN_P0_BIH), GiP0, 768, PT, 768, 832);
  k_gru_seq<<<2, 512, 0, stream>>>(GiP0, 0, p0Whh, F(IN_P0_BHH), hsPa, (size_t)PT * 256, 256, 0, 256);
  gemm(hsPa, 256, p1Wih, 256, F(IN_P1_BIH), GiPd, 768, PT, 768, 256);
  gemm(hsPa + (size_t)PT * 256, 256, p1Wih, 256, F(IN_P1_BIH), GiPd + (size_t)PT * 768, 768, PT, 768, 256);
  k_gru_seq<<<2, 512, 0, stream>>>(GiPd, (size_t)PT * 768, p1Whh, F(IN_P1_BHH), hsPb, (size_t)PT * 256, 256, 0, 256);
  gemm(hsPb, 256, p2Wih, 256, F(IN_P2_BIH), GiPd, 768, PT, 768, 256);
  gemm(hsPb + (size_t)PT * 256, 256, p2Wih, 256, F(IN_P2_BIH), GiPd + (size_t)PT * 768, 768, PT, 768, 256);
  k_gru_seq<<<2, 512, 0, stream>>>(GiPd, (size_t)PT * 768, p2Whh, F(IN_P2_BHH), HpF, 0, 512, 256, 256);

  // ---- question 3-layer bi-GRU ----
  gemm(qx, 832, p0Wih, 832, F(IN_P0_BIH), GiQ0, 768, QT, 768, 832);
  k_gru_seq<<<2, 512, 0, stream>>>(GiQ0, 0, p0Whh, F(IN_P0_BHH), hsQa, (size_t)QT * 256, 256, 0, 30);
  gemm(hsQa, 256, p1Wih, 256, F(IN_P1_BIH), GiQd, 768, QT, 768, 256);
  gemm(hsQa + (size_t)QT * 256, 256, p1Wih, 256, F(IN_P1_BIH), GiQd + (size_t)QT * 768, 768, QT, 768, 256);
  k_gru_seq<<<2, 512, 0, stream>>>(GiQd, (size_t)QT * 768, p1Whh, F(IN_P1_BHH), hsQb, (size_t)QT * 256, 256, 0, 30);
  gemm(hsQb, 256, p2Wih, 256, F(IN_P2_BIH), GiQd, 768, QT, 768, 256);
  gemm(hsQb + (size_t)QT * 256, 256, p2Wih, 256, F(IN_P2_BIH), GiQd + (size_t)QT * 768, 768, QT, 768, 256);
  k_gru_seq<<<2, 512, 0, stream>>>(GiQd, (size_t)QT * 768, p2Whh, F(IN_P2_BHH), HqF, 0, 512, 256, 30);

  // ---- match-LSTM (bidirectional) ----
  gemm(HqF, 512, aQ, 512, nullptr, preQ, 256, QT, 256, 512);
  gemm(HpF, 512, aP, 512, nullptr, hinWP, 256, PT, 256, 512);
  k_attn_lstm<<<2, 512, 0, stream>>>(preQ, hinWP, HqF, HpF, aH, F(IN_ALPHA), gM, mlWih, mlWhh,
                                     F(IN_ML_BIH), F(IN_ML_BHH), Hm, 512, 256, 256, 30);

  // ---- self-matching (bidirectional) ----
  gemm(Hm, 512, aSP, 512, nullptr, preS, 256, PT, 256, 512);
  k_attn_lstm<<<2, 512, 0, stream>>>(preS, nullptr, Hm, Hm, aSH, F(IN_GAMMA), gS, slWih, slWhh,
                                     F(IN_SL_BIH), F(IN_SL_BHH), Hs, 512, 256, 256, 256);

  // ---- answer pointer ----
  gemm(Hs, 512, aSL, 512, nullptr, preA, 256, PT, 256, 512);
  k_ans<<<1, 512, 0, stream>>>(preA, Hs, aANS, F(IN_BETA), anWih, anWhh, F(IN_ANS_BIH),
                               F(IN_ANS_BHH), (float*)d_out);
}
